// EfficientSparseCODMoE_77232101916873
// MI455X (gfx1250) — matrix operand order repaired and checked by reading, not compiled, not run on hardware
//
#include <hip/hip_runtime.h>
#include <cstdint>
#include <cstddef>

// ---------------------------------------------------------------------------
// MoE conv (top-2 of 6 experts), bf16x3-split implicit-GEMM on CDNA5 WMMA.
//   x:[8,64,128,128] f32 NCHW ; experts: conv3x3 -> GELU -> conv3x3 (64->64)
// Only the 2 routed experts per sample are computed (1/3 of reference work).
// Precision: fp32 emulated as w_hi*x_hi + w_hi*x_lo + w_lo*x_hi (bf16 parts,
// f32 accumulate in v_wmma_f32_16x16x32_bf16) -> K_eff = 3*576 = 1728.
// LDS halo staging uses global_load_async_to_lds_b128 (ASYNCcnt-tracked).
// ---------------------------------------------------------------------------

typedef __attribute__((ext_vector_type(16))) __bf16 v16bf;
typedef __attribute__((ext_vector_type(8)))  float  v8f;

union ABPack {
    uint32_t u[8];
    uint4    q[2];
    v16bf    v;
};

__device__ __forceinline__ unsigned short f2bf(float f) {
    uint32_t u = __float_as_uint(f);
    u += 0x7FFFu + ((u >> 16) & 1u);      // round-to-nearest-even
    return (unsigned short)(u >> 16);
}
__device__ __forceinline__ float bf2f(unsigned short h) {
    return __uint_as_float(((uint32_t)h) << 16);
}
__device__ __forceinline__ float gelu_f(float x) {
    // jax.nn.gelu default (approximate=True, tanh form)
    float x3 = x * x * x;
    return 0.5f * x * (1.0f + tanhf(0.7978845608028654f * (x + 0.044715f * x3)));
}

#define HW_PIX     16384      // 128*128
#define CDIM       64
#define NCHUNK     54         // K_eff/32 = 1728/32
#define APACK_DW   663552     // 12 * 54 * 4 * 32 * 8 dwords
#define PLANE_US   12672      // 3*66*64 ushorts per LDS plane

// ---------------------------------------------------------------------------
// 1) Split x (NCHW f32) -> NHWC bf16 hi/lo planes; deterministic partial sums
//    for the router's global average pool (one partial per 256-thread block).
// ---------------------------------------------------------------------------
__global__ __launch_bounds__(256)
void k_split_pool(const float* __restrict__ x,
                  unsigned short* __restrict__ xhi,
                  unsigned short* __restrict__ xlo,
                  float* __restrict__ partial) {
    int gid = blockIdx.x * 256 + threadIdx.x;           // 2,097,152 total
    float v = x[gid];
    unsigned short hi = f2bf(v);
    float lo = v - bf2f(hi);
    int p  = gid & (HW_PIX - 1);
    int bc = gid >> 14;                                  // b*64 + c
    int b  = bc >> 6, c = bc & 63;
    size_t dst = ((size_t)(b * HW_PIX + p)) * CDIM + c;  // NHWC
    xhi[dst] = hi;
    xlo[dst] = f2bf(lo);

    __shared__ float red[256];
    red[threadIdx.x] = v;
    __syncthreads();
    for (int s = 128; s > 0; s >>= 1) {
        if (threadIdx.x < s) red[threadIdx.x] += red[threadIdx.x + s];
        __syncthreads();
    }
    if (threadIdx.x == 0) partial[blockIdx.x] = red[0];  // blockIdx = bc*64 + j
}

// ---------------------------------------------------------------------------
// 2) Pack expert weights into WMMA A-operand layout (16x32 bf16 tile layout,
//    ISA 7.12.2). K order: term*576 + (kh*3+kw)*64 + ci. term0,1 -> w_hi,
//    term2 -> w_lo.  Apack dword layout: [ew][chunk][mt][lane][8].
// ---------------------------------------------------------------------------
__global__ __launch_bounds__(256)
void k_pack_weights(const float* __restrict__ w1, const float* __restrict__ w2,
                    uint32_t* __restrict__ Apack) {
    int gid = blockIdx.x * 256 + threadIdx.x;
    if (gid >= APACK_DW) return;
    int d    = gid & 7;  int tmp = gid >> 3;
    int lane = tmp & 31; tmp >>= 5;
    int mt   = tmp & 3;  tmp >>= 2;
    int c    = tmp % NCHUNK;
    int ew   = tmp / NCHUNK;                 // expert*2 + conv
    int e    = ew >> 1, cv = ew & 1;
    const float* W = (cv ? w2 : w1) + (size_t)e * 64 * 576;  // OIHW flat
    int half = lane >> 4, m = lane & 15;
    int co = mt * 16 + m;
    // A layout: lane<16: V0-3->K0..7, V4-7->K16..23 ; lane>=16: +8
    int K0 = (d & 3) * 2 + (d >> 2) * 16 + half * 8;     // even
    int k0 = c * 32 + K0;
    int t3 = k0 / 576, r = k0 % 576;
    int khkw = r >> 6, ci = r & 63;
    float f0 = W[(size_t)co * 576 + ci * 9 + khkw];
    float f1 = W[(size_t)co * 576 + (ci + 1) * 9 + khkw];
    unsigned short h0 = f2bf(f0), h1 = f2bf(f1);
    unsigned short r0 = (t3 < 2) ? h0 : f2bf(f0 - bf2f(h0));
    unsigned short r1 = (t3 < 2) ? h1 : f2bf(f1 - bf2f(h1));
    Apack[(size_t)gid] = (uint32_t)r0 | ((uint32_t)r1 << 16);
}

// ---------------------------------------------------------------------------
// 3) Router: deterministic pool reduce -> linear -> clamp -> +bias -> softmax
//    -> clip -> top-2 (first-index ties, like lax.top_k) -> renormalize.
// ---------------------------------------------------------------------------
__global__ __launch_bounds__(512)
void k_router(const float* __restrict__ partial,
              const float* __restrict__ rw, const float* __restrict__ rb,
              const float* __restrict__ eb,
              int* __restrict__ gate_idx, float* __restrict__ gate_w,
              float* __restrict__ out_scalar) {
    __shared__ float pooled[512];
    __shared__ float logits[48];
    int t = threadIdx.x;
    {
        float s = 0.f;
        const float* p = partial + (size_t)t * 64;
        for (int j = 0; j < 64; ++j) s += p[j];
        pooled[t] = s * (1.0f / 16384.0f);
    }
    __syncthreads();
    if (t < 48) {
        int b = t / 6, ei = t % 6;
        float s = 0.f;
        for (int c = 0; c < 64; ++c) s += pooled[b * 64 + c] * rw[ei * 64 + c];
        s += rb[ei];
        s = fminf(10.0f, fmaxf(-10.0f, s));
        s += eb[ei];
        logits[t] = s;
    }
    __syncthreads();
    if (t < 8) {
        int b = t;
        float m = -1e30f;
        for (int i = 0; i < 6; ++i) m = fmaxf(m, logits[b * 6 + i]);
        float pe[6]; float sum = 0.f;
        for (int i = 0; i < 6; ++i) { pe[i] = expf(logits[b * 6 + i] - m); sum += pe[i]; }
        for (int i = 0; i < 6; ++i) pe[i] = fminf(1.0f, fmaxf(1e-6f, pe[i] / sum));
        int i0 = 0; float v0 = pe[0];
        for (int i = 1; i < 6; ++i) if (pe[i] > v0) { v0 = pe[i]; i0 = i; }
        int i1 = -1; float v1 = -1.f;
        for (int i = 0; i < 6; ++i) if (i != i0 && pe[i] > v1) { v1 = pe[i]; i1 = i; }
        float s2 = v0 + v1 + 1e-8f;
        gate_idx[b * 2 + 0] = i0;  gate_idx[b * 2 + 1] = i1;
        gate_w [b * 2 + 0] = v0 / s2;  gate_w [b * 2 + 1] = v1 / s2;
    }
    if (t == 0) *out_scalar = 0.0f;
}

// ---------------------------------------------------------------------------
// 4) Implicit-GEMM 3x3 conv via WMMA bf16x3. One block = 64 co x 64 pixels.
//    8 waves = 4(M) x 2(N-group); each wave holds two 16x16 f32 accumulators.
//    3x66 halo x 64ci staged in LDS (hi+lo planes, one array) via
//    global_load_async_to_lds_b128; padding zero-filled with ds stores.
//    K loop fully unrolled: all A/B offsets become instruction immediates.
// ---------------------------------------------------------------------------
__global__ __launch_bounds__(256)
void k_moe_conv(const unsigned short* __restrict__ shi,
                const unsigned short* __restrict__ slo,
                const uint32_t* __restrict__ Apack,
                const float* __restrict__ bias,           // [6][64] base
                const int* __restrict__ gate_idx, const float* __restrict__ gate_w,
                int bsample, int slot, int conv_idx,
                unsigned short* __restrict__ dhi, unsigned short* __restrict__ dlo,
                float* __restrict__ outp) {
    // plane 0 = hi, plane 1 = lo ; each plane 3*66*64 ushorts
    __shared__ __align__(16) unsigned short s_data[2 * PLANE_US];

    const int tid = threadIdx.x;
    const int y   = blockIdx.x >> 1;
    const int x0  = (blockIdx.x & 1) << 6;
    const int e   = gate_idx[bsample * 2 + slot];
    const float g = gate_w [bsample * 2 + slot];
    const int ew  = e * 2 + conv_idx;

    const int wave = tid >> 5, lane = tid & 31;
    const int mt = wave >> 1, ng = wave & 1;
    const int half = lane >> 4, n = lane & 15;

    // A stream base for this wave (chunk stride = 256 uint4 = 4096 B)
    const uint4* Aq = (const uint4*)Apack +
                      (size_t)(((ew * (NCHUNK * 4) + mt) * 32) + lane) * 2;
    __builtin_prefetch((const void*)Aq, 0, 1);           // warm L2 during staging

    // --- stage 3x66 halo (hi & lo planes) via async global->LDS copies ---
    for (int t = tid; t < 396 * 8; t += 256) {
        int unit = t >> 3, sub = t & 7;
        int pl = (unit >= 198) ? 1 : 0;
        int u  = unit - pl * 198;
        int rr = u / 66, cc = u - rr * 66;
        int gy = y - 1 + rr, gx = x0 - 1 + cc;
        unsigned short* ldst = s_data + (size_t)pl * PLANE_US +
                               (size_t)((rr * 66 + cc) * 64 + sub * 8);
        if (gy >= 0 && gy < 128 && gx >= 0 && gx < 128) {
            const unsigned short* src = pl ? slo : shi;
            const unsigned short* gsrc = src + ((size_t)(gy * 128 + gx) * 64 + sub * 8);
            uint32_t loff = (uint32_t)(uintptr_t)ldst;   // LDS aperture: low 32 = offset
            uint64_t ga   = (uint64_t)(uintptr_t)gsrc;
            asm volatile("global_load_async_to_lds_b128 %0, %1, off"
                         :: "v"(loff), "v"(ga) : "memory");
        } else {
            *(uint4*)ldst = make_uint4(0u, 0u, 0u, 0u);  // 'SAME' padding
        }
    }
    asm volatile("s_wait_asynccnt 0x0" ::: "memory");
    __syncthreads();

    // B operand lane base: column = ng*32 + n, K layout half*8 + {0..7, 16..23}
    const unsigned short* Bbase = s_data + (size_t)(((ng << 5) + n) * 64) + (half << 3);

    v8f acc0 = {0.f,0.f,0.f,0.f,0.f,0.f,0.f,0.f};
    v8f acc1 = {0.f,0.f,0.f,0.f,0.f,0.f,0.f,0.f};

    // Fully unrolled K loop: chunk = t3*18 + (kh*3+kw)*2 + ic.  Every offset
    // below is a compile-time constant -> folds into load immediates.
#pragma unroll
    for (int t3 = 0; t3 < 3; ++t3) {
#pragma unroll
        for (int khkw = 0; khkw < 9; ++khkw) {
            const int kh = khkw / 3, kw = khkw % 3;
#pragma unroll
            for (int ic = 0; ic < 2; ++ic) {
                const int cidx = t3 * 18 + khkw * 2 + ic;
                const int boff = (t3 == 1 ? PLANE_US : 0) +
                                 (kh * 66 + kw) * 64 + (ic << 5);
                ABPack a;
                a.q[0] = Aq[cidx * 256 + 0];
                a.q[1] = Aq[cidx * 256 + 1];
                ABPack bm0, bm1;
                bm0.q[0] = *(const uint4*)(Bbase + boff);
                bm0.q[1] = *(const uint4*)(Bbase + boff + 16);
                bm1.q[0] = *(const uint4*)(Bbase + boff + 1024);       // +16 px
                bm1.q[1] = *(const uint4*)(Bbase + boff + 1040);
                acc0 = __builtin_amdgcn_wmma_f32_16x16x32_bf16(
                           false, a.v, false, bm0.v, (short)0, acc0, false, false);
                acc1 = __builtin_amdgcn_wmma_f32_16x16x32_bf16(
                           false, a.v, false, bm1.v, (short)0, acc1, false, false);
            }
        }
    }

    // C/D layout: lane l, VGPR r -> co = mt*16 + 8*(l/16) + r, pixel = l%16
    const int coB = mt * 16 + half * 8;
    float bloc[8];
    for (int r = 0; r < 8; ++r) bloc[r] = bias[e * 64 + coB + r];

    if (conv_idx == 0) {
        for (int s = 0; s < 2; ++s) {
            v8f acc = s ? acc1 : acc0;
            int col = x0 + (ng << 5) + (s << 4) + n;
            union { unsigned short us[8]; uint4 q; } ph, pl2;
            for (int r = 0; r < 8; ++r) {
                float v = acc[r] + bloc[r];
                float ge = gelu_f(v);
                unsigned short hi = f2bf(ge);
                ph.us[r]  = hi;
                pl2.us[r] = f2bf(ge - bf2f(hi));
            }
            size_t base = (size_t)(y * 128 + col) * 64 + coB;
            *(uint4*)(dhi + base) = ph.q;
            *(uint4*)(dlo + base) = pl2.q;
        }
    } else {
        for (int s = 0; s < 2; ++s) {
            v8f acc = s ? acc1 : acc0;
            int col = x0 + (ng << 5) + (s << 4) + n;
            for (int r = 0; r < 8; ++r) {
                float v = (acc[r] + bloc[r]) * g;
                size_t oidx = (((size_t)bsample * 64 + coB + r) << 14) +
                              (size_t)(y * 128 + col);
                if (slot) outp[oidx] += v;   // slot1 accumulates (stream-ordered)
                else      outp[oidx]  = v;   // slot0 overwrites poisoned d_out
            }
        }
    }
}

// ---------------------------------------------------------------------------
extern "C" void kernel_launch(void* const* d_in, const int* in_sizes, int n_in,
                              void* d_out, int out_size, void* d_ws, size_t ws_size,
                              hipStream_t stream) {
    (void)in_sizes; (void)n_in; (void)out_size; (void)ws_size;
    const float* x           = (const float*)d_in[0];
    const float* router_w    = (const float*)d_in[1];
    const float* router_b    = (const float*)d_in[2];
    const float* expert_bias = (const float*)d_in[3];
    const float* w1          = (const float*)d_in[4];
    const float* b1          = (const float*)d_in[5];
    const float* w2          = (const float*)d_in[6];
    const float* b2          = (const float*)d_in[7];
    float* out = (float*)d_out;

    // ---- workspace carve (all regions fully rewritten each call) ----
    auto align_up = [](size_t v) { return (v + 255) & ~(size_t)255; };
    char* ws = (char*)d_ws;
    size_t off = 0;
    float* partial = (float*)(ws + off);          off = align_up(off + 8192 * 4);
    int*   gate_idx = (int*)(ws + off);           off = align_up(off + 16 * 4);
    float* gate_w   = (float*)(ws + off);         off = align_up(off + 16 * 4);
    uint32_t* Apack = (uint32_t*)(ws + off);      off = align_up(off + (size_t)APACK_DW * 4);
    unsigned short* xhi = (unsigned short*)(ws + off); off = align_up(off + (size_t)8 * HW_PIX * CDIM * 2);
    unsigned short* xlo = (unsigned short*)(ws + off); off = align_up(off + (size_t)8 * HW_PIX * CDIM * 2);
    unsigned short* hhi = (unsigned short*)(ws + off); off = align_up(off + (size_t)HW_PIX * CDIM * 2);
    unsigned short* hlo = (unsigned short*)(ws + off); off = align_up(off + (size_t)HW_PIX * CDIM * 2);

    k_split_pool<<<8192, 256, 0, stream>>>(x, xhi, xlo, partial);
    k_pack_weights<<<(APACK_DW + 255) / 256, 256, 0, stream>>>(w1, w2, Apack);
    k_router<<<1, 512, 0, stream>>>(partial, router_w, router_b, expert_bias,
                                    gate_idx, gate_w, out + (size_t)8 * CDIM * HW_PIX);

    // Stream-serialized (b, slot) pairs share one h buffer safely.
    for (int b = 0; b < 8; ++b) {
        for (int slot = 0; slot < 2; ++slot) {
            const unsigned short* sx_hi = xhi + (size_t)b * HW_PIX * CDIM;
            const unsigned short* sx_lo = xlo + (size_t)b * HW_PIX * CDIM;
            k_moe_conv<<<256, 256, 0, stream>>>(sx_hi, sx_lo, Apack, b1,
                                                gate_idx, gate_w, b, slot, 0,
                                                hhi, hlo, nullptr);
            k_moe_conv<<<256, 256, 0, stream>>>(hhi, hlo, Apack, b2,
                                                gate_idx, gate_w, b, slot, 1,
                                                nullptr, nullptr, out);
        }
    }
}
